// AttnHead_21363167330660
// MI455X (gfx1250) — matrix operand order repaired and checked by reading, not compiled, not run on hardware
//
#include <hip/hip_runtime.h>

typedef __attribute__((ext_vector_type(16))) _Float16 v16h;
typedef __attribute__((ext_vector_type(8)))  float    v8f;
typedef __attribute__((ext_vector_type(4)))  float    v4f;

#define NTOK  8192
#define FIN   256
#define FOUT  64
#define PARTS 8
#define CPP   (NTOK / PARTS)   // 1024 columns per partition
#define ST    64               // columns per supertile (two 16x32 A tiles)

// ---------------------------------------------------------------------------
// Kernel 1: seq_fts = seq @ W1  (8192x256 @ 256x64), f32 result + f16 transpose
// ---------------------------------------------------------------------------
__global__ __launch_bounds__(256) void gat_gemm1(const float* __restrict__ seq,
                                                 const float* __restrict__ W1,
                                                 float* __restrict__ sf,
                                                 _Float16* __restrict__ sfT) {
  const int tid = threadIdx.x;
  const int row = blockIdx.x * 32 + (tid >> 3);   // 32 rows per block
  const int cg  = (tid & 7) * 8;                  // 8 output cols per thread
  const float* srow = seq + (size_t)row * FIN;
  float acc[8];
#pragma unroll
  for (int j = 0; j < 8; ++j) acc[j] = 0.f;
  for (int k = 0; k < FIN; k += 4) {
    v4f sv = *(const v4f*)(srow + k);
#pragma unroll
    for (int kk = 0; kk < 4; ++kk) {
      const float a = sv[kk];
      v4f w0 = *(const v4f*)(W1 + (size_t)(k + kk) * FOUT + cg);
      v4f w1 = *(const v4f*)(W1 + (size_t)(k + kk) * FOUT + cg + 4);
#pragma unroll
      for (int j = 0; j < 4; ++j) { acc[j] += a * w0[j]; acc[4 + j] += a * w1[j]; }
    }
  }
#pragma unroll
  for (int j = 0; j < 8; ++j) {
    sf[(size_t)row * FOUT + cg + j]    = acc[j];
    sfT[(size_t)(cg + j) * NTOK + row] = (_Float16)acc[j];
  }
}

// ---------------------------------------------------------------------------
// Kernel 2: f1 = seq_fts @ a1 + b1 ; f2 = seq_fts @ a2 + b2  (per row)
// ---------------------------------------------------------------------------
__global__ __launch_bounds__(256) void gat_fvec(const float* __restrict__ sf,
                                                const float* __restrict__ a1,
                                                const float* __restrict__ b1,
                                                const float* __restrict__ a2,
                                                const float* __restrict__ b2,
                                                float* __restrict__ f1,
                                                float* __restrict__ f2) {
  const int r = blockIdx.x * blockDim.x + threadIdx.x;
  const float* rp = sf + (size_t)r * FOUT;
  float s1 = 0.f, s2 = 0.f;
#pragma unroll 8
  for (int k = 0; k < FOUT; ++k) { const float v = rp[k]; s1 += v * a1[k]; s2 += v * a2[k]; }
  f1[r] = s1 + b1[0];
  f2[r] = s2 + b2[0];
}

// ---------------------------------------------------------------------------
// Kernel 3: fused flash-softmax + (coefs @ seq_fts) via v_wmma_f32_16x16x32_f16
// One wave = one 16-row block x one 1024-column partition.
// Supertile = 64 columns = two A tiles; ping-pong software pipeline on the
// bias + B-tile global loads so each wave hides a full iteration of latency.
// A layout: lane l -> M=l%16, element e -> K = e + 16*(l>=16)
// B layout: lane l -> N=l%16, element e -> K = e + 16*(l>=16) (from sfT rows)
// C layout: VGPR r, lane l -> M = r + 8*(l>=16), N = l%16
// ---------------------------------------------------------------------------
__global__ __launch_bounds__(128) void gat_flash(const float* __restrict__ bias,
                                                 const float* __restrict__ f1v,
                                                 const float* __restrict__ f2v,
                                                 const _Float16* __restrict__ sfT,
                                                 float* __restrict__ accP,
                                                 float* __restrict__ mP,
                                                 float* __restrict__ lP) {
  const int lane = threadIdx.x & 31;
  const int wave = threadIdx.x >> 5;
  const int blk  = blockIdx.x;                 // 0..511 row block
  const int part = blockIdx.y * 4 + wave;      // 0..7 column partition
  const int h    = lane >> 4;                  // half-wave select
  const int M    = lane & 15;                  // this lane's attention row
  const int c0   = part * CPP;
  const int cend = c0 + CPP;

  const float f1 = f1v[blk * 16 + M];
  const float* brow = bias + (size_t)(blk * 16 + M) * NTOK;
  const _Float16* sb = sfT + (size_t)M * NTOK + 16 * h;

  float mrow = -3.0e38f, lrow = 0.f;
  v8f acc0 = {}, acc1 = {}, acc2 = {}, acc3 = {};

  // load one 64-column supertile of bias (A-layout runs) and B tiles
  auto loadt = [&](int cb, v4f (&bb)[8], v16h (&B)[8]) {
    const float* p = brow + cb + 16 * h;
    bb[0] = *(const v4f*)(p);       bb[1] = *(const v4f*)(p + 4);
    bb[2] = *(const v4f*)(p + 8);   bb[3] = *(const v4f*)(p + 12);
    bb[4] = *(const v4f*)(p + 32);  bb[5] = *(const v4f*)(p + 36);
    bb[6] = *(const v4f*)(p + 40);  bb[7] = *(const v4f*)(p + 44);
    const _Float16* q = sb + cb;
#pragma unroll
    for (int t = 0; t < 4; ++t) {
      B[t]     = *(const v16h*)(q + (size_t)t * 16 * NTOK);
      B[4 + t] = *(const v16h*)(q + (size_t)t * 16 * NTOK + 32);
    }
  };

  auto process = [&](int cb, const v4f (&bb)[8], const v16h (&B)[8]) {
    const float* fp = f2v + cb + 16 * h;           // tiny, L1/L2-hot
    v4f ff[8];
    ff[0] = *(const v4f*)(fp);       ff[1] = *(const v4f*)(fp + 4);
    ff[2] = *(const v4f*)(fp + 8);   ff[3] = *(const v4f*)(fp + 12);
    ff[4] = *(const v4f*)(fp + 32);  ff[5] = *(const v4f*)(fp + 36);
    ff[6] = *(const v4f*)(fp + 40);  ff[7] = *(const v4f*)(fp + 44);

    float s[32];
#pragma unroll
    for (int q = 0; q < 8; ++q)
#pragma unroll
      for (int j = 0; j < 4; ++j) {
        float x = f1 + ff[q][j];
        s[q * 4 + j] = fmaxf(x, 0.2f * x) + bb[q][j];   // leaky_relu(.,0.2)+bias
      }

    // 4-way max tree (short dependency chains)
    float tm0 = fmaxf(s[0], s[4]),  tm1 = fmaxf(s[1], s[5]);
    float tm2 = fmaxf(s[2], s[6]),  tm3 = fmaxf(s[3], s[7]);
#pragma unroll
    for (int e = 8; e < 32; e += 4) {
      tm0 = fmaxf(tm0, s[e]);     tm1 = fmaxf(tm1, s[e + 1]);
      tm2 = fmaxf(tm2, s[e + 2]); tm3 = fmaxf(tm3, s[e + 3]);
    }
    float tmax = fmaxf(fmaxf(tm0, tm1), fmaxf(tm2, tm3));
    tmax = fmaxf(tmax, __shfl_xor(tmax, 16, 32));        // pair lanes share row M
    const float mnew  = fmaxf(mrow, tmax);
    const float alpha = __expf(mrow - mnew);

    v16h a0, a1;
    float p0s = 0.f, p1s = 0.f, p2s = 0.f, p3s = 0.f;
#pragma unroll
    for (int e = 0; e < 16; e += 2) {
      float pa = __expf(s[e] - mnew);
      float pb = __expf(s[e + 1] - mnew);
      float pc = __expf(s[16 + e] - mnew);
      float pd = __expf(s[17 + e] - mnew);
      p0s += pa; p1s += pb; p2s += pc; p3s += pd;
      a0[e] = (_Float16)pa; a0[e + 1] = (_Float16)pb;
      a1[e] = (_Float16)pc; a1[e + 1] = (_Float16)pd;
    }
    float psum = (p0s + p1s) + (p2s + p3s);
    psum += __shfl_xor(psum, 16, 32);
    lrow = lrow * alpha + psum;
    mrow = mnew;

    // broadcast per-row rescale into C layout (row = r + 8*h)
    float asc[8];
#pragma unroll
    for (int r = 0; r < 8; ++r) asc[r] = __shfl(alpha, r + 8 * h, 32);
#pragma unroll
    for (int r = 0; r < 8; ++r) {
      acc0[r] *= asc[r]; acc1[r] *= asc[r]; acc2[r] *= asc[r]; acc3[r] *= asc[r];
    }

    acc0 = __builtin_amdgcn_wmma_f32_16x16x32_f16(false, a0, false, B[0], (short)0, acc0, false, false);
    acc1 = __builtin_amdgcn_wmma_f32_16x16x32_f16(false, a0, false, B[1], (short)0, acc1, false, false);
    acc2 = __builtin_amdgcn_wmma_f32_16x16x32_f16(false, a0, false, B[2], (short)0, acc2, false, false);
    acc3 = __builtin_amdgcn_wmma_f32_16x16x32_f16(false, a0, false, B[3], (short)0, acc3, false, false);
    acc0 = __builtin_amdgcn_wmma_f32_16x16x32_f16(false, a1, false, B[4], (short)0, acc0, false, false);
    acc1 = __builtin_amdgcn_wmma_f32_16x16x32_f16(false, a1, false, B[5], (short)0, acc1, false, false);
    acc2 = __builtin_amdgcn_wmma_f32_16x16x32_f16(false, a1, false, B[6], (short)0, acc2, false, false);
    acc3 = __builtin_amdgcn_wmma_f32_16x16x32_f16(false, a1, false, B[7], (short)0, acc3, false, false);
  };

  v4f bbA[8], bbB[8];
  v16h BA[8], BB[8];
  loadt(c0, bbA, BA);
  for (int cb = c0; cb < cend; cb += 2 * ST) {
    // hint-prefetch the bias row ~4KB ahead (global_prefetch path)
    const int pf = cb + 1024;
    if (pf < NTOK) __builtin_prefetch(brow + pf, 0, 1);
    loadt(cb + ST, bbB, BB);
    process(cb, bbA, BA);
    const int nxt = (cb + 2 * ST < cend) ? (cb + 2 * ST) : c0;  // wrap: dead load on last iter
    loadt(nxt, bbA, BA);
    process(cb + ST, bbB, BB);
  }

  const size_t pb = (size_t)(blk * PARTS + part);
  if (h == 0) { mP[pb * 16 + M] = mrow; lP[pb * 16 + M] = lrow; }
  float* ap = accP + pb * 1024;
#pragma unroll
  for (int r = 0; r < 8; ++r) {
    ap[0 * 256 + r * 32 + lane] = acc0[r];
    ap[1 * 256 + r * 32 + lane] = acc1[r];
    ap[2 * 256 + r * 32 + lane] = acc2[r];
    ap[3 * 256 + r * 32 + lane] = acc3[r];
  }
}

// ---------------------------------------------------------------------------
// Kernel 4: merge the 8 column-partition partials per row block, ELU, write out
// ---------------------------------------------------------------------------
__global__ __launch_bounds__(32) void gat_combine(const float* __restrict__ accP,
                                                  const float* __restrict__ mP,
                                                  const float* __restrict__ lP,
                                                  float* __restrict__ out) {
  const int lane = threadIdx.x & 31;
  const int h    = lane >> 4;
  const int nn   = lane & 15;
  const int blk  = blockIdx.x;

  float mr[8];
#pragma unroll
  for (int r = 0; r < 8; ++r) mr[r] = -3.0e38f;
  for (int p = 0; p < PARTS; ++p) {
    const float* mpp = mP + (size_t)(blk * PARTS + p) * 16;
#pragma unroll
    for (int r = 0; r < 8; ++r) mr[r] = fmaxf(mr[r], mpp[r + 8 * h]);
  }

  float den[8];
  v8f n0 = {}, n1 = {}, n2 = {}, n3 = {};
#pragma unroll
  for (int r = 0; r < 8; ++r) den[r] = 0.f;

  for (int p = 0; p < PARTS; ++p) {
    const float* mpp = mP + (size_t)(blk * PARTS + p) * 16;
    const float* lpp = lP + (size_t)(blk * PARTS + p) * 16;
    const float* app = accP + (size_t)(blk * PARTS + p) * 1024;
    float w[8];
#pragma unroll
    for (int r = 0; r < 8; ++r) {
      w[r] = __expf(mpp[r + 8 * h] - mr[r]);
      den[r] += lpp[r + 8 * h] * w[r];
    }
#pragma unroll
    for (int r = 0; r < 8; ++r) {
      n0[r] += app[0 * 256 + r * 32 + lane] * w[r];
      n1[r] += app[1 * 256 + r * 32 + lane] * w[r];
      n2[r] += app[2 * 256 + r * 32 + lane] * w[r];
      n3[r] += app[3 * 256 + r * 32 + lane] * w[r];
    }
  }
#pragma unroll
  for (int r = 0; r < 8; ++r) {
    const size_t ob = (size_t)(blk * 16 + r + 8 * h) * FOUT + nn;
    float v0 = n0[r] / den[r]; out[ob + 0]  = v0 > 0.f ? v0 : expm1f(v0);
    float v1 = n1[r] / den[r]; out[ob + 16] = v1 > 0.f ? v1 : expm1f(v1);
    float v2 = n2[r] / den[r]; out[ob + 32] = v2 > 0.f ? v2 : expm1f(v2);
    float v3 = n3[r] / den[r]; out[ob + 48] = v3 > 0.f ? v3 : expm1f(v3);
  }
}

// ---------------------------------------------------------------------------
// Launch
// ---------------------------------------------------------------------------
extern "C" void kernel_launch(void* const* d_in, const int* in_sizes, int n_in,
                              void* d_out, int out_size, void* d_ws, size_t ws_size,
                              hipStream_t stream) {
  const float* seq  = (const float*)d_in[0];
  const float* bias = (const float*)d_in[1];
  const float* W1   = (const float*)d_in[2];
  const float* a1   = (const float*)d_in[3];
  const float* b1   = (const float*)d_in[4];
  const float* a2   = (const float*)d_in[5];
  const float* b2   = (const float*)d_in[6];
  float* out = (float*)d_out;

  char* ws = (char*)d_ws;
  float*    sf   = (float*)(ws);                 // seq_fts f32   : 2,097,152 B
  _Float16* sfT  = (_Float16*)(ws + 2097152);    // seq_fts^T f16 : 1,048,576 B
  float*    f1   = (float*)(ws + 3145728);       // 8192*4
  float*    f2   = (float*)(ws + 3178496);       // 8192*4
  float*    mP   = (float*)(ws + 3211264);       // 262,144 B
  float*    lP   = (float*)(ws + 3473408);       // 262,144 B
  float*    accP = (float*)(ws + 3735552);       // 16,777,216 B

  hipLaunchKernelGGL(gat_gemm1,   dim3(NTOK / 32),    dim3(256), 0, stream, seq, W1, sf, sfT);
  hipLaunchKernelGGL(gat_fvec,    dim3(NTOK / 256),   dim3(256), 0, stream, sf, a1, b1, a2, b2, f1, f2);
  hipLaunchKernelGGL(gat_flash,   dim3(NTOK / 16, 2), dim3(128), 0, stream, bias, f1, f2, sfT, accP, mP, lP);
  hipLaunchKernelGGL(gat_combine, dim3(NTOK / 16),    dim3(32),  0, stream, accP, mP, lP, out);
}
